// NCWNO2d_21328807592356
// MI455X (gfx1250) — matrix operand order repaired and checked by reading, not compiled, not run on hardware
//
#include <hip/hip_runtime.h>
#include <math.h>

// ---------------- problem constants ----------------
#define NB   16          // batch
#define CC   64          // channels
#define HWP  16384       // 128*128 pixels
#define HID  4

typedef __attribute__((ext_vector_type(16))) __bf16 v16bf;
typedef __attribute__((ext_vector_type(8)))  __bf16 v8bf;
typedef __attribute__((ext_vector_type(8)))  float  v8f;

union ABfrag { v16bf v; v8bf h[2]; };

__device__ __forceinline__ unsigned short f2bf(float f) {
  unsigned int u = __float_as_uint(f);
  u += 0x7FFFu + ((u >> 16) & 1u);          // round-to-nearest-even
  return (unsigned short)(u >> 16);
}

// mish(x) = x * tanh(softplus(x));  tanh(log(1+e^x)) = ((1+e^x)^2-1)/((1+e^x)^2+1)
__device__ __forceinline__ float mishf(float x) {
  float t = __expf(fminf(x, 32.0f));
  float s = 1.0f + t;
  float u = s * s;
  float r = x * (u - 1.0f) / (u + 1.0f);
  return (x > 32.0f) ? x : r;
}

// stage 16 pixels x 64 channels (f32 -> bf16) into LDS [M=16][K=64]
// in points at (b, c=0, p0); channel stride = STRIDE pixels.
template <int STRIDE>
__device__ __forceinline__ void stageA(const float* __restrict__ in,
                                       unsigned short* __restrict__ ldsA,
                                       int lane) {
#pragma unroll
  for (int it = 0; it < 8; ++it) {
    int idx = lane + it * 32;        // 256 chunks of 4 pixels
    int c = idx >> 2, m4 = (idx & 3) * 4;
    const float4 f = *(const float4*)(in + (size_t)c * STRIDE + m4);
    ldsA[(m4 + 0) * 64 + c] = f2bf(f.x);
    ldsA[(m4 + 1) * 64 + c] = f2bf(f.y);
    ldsA[(m4 + 2) * 64 + c] = f2bf(f.z);
    ldsA[(m4 + 3) * 64 + c] = f2bf(f.w);
  }
}

// ---------------- weight fragment packing ----------------
// frag layout: ((nt*2 + kt)*32 + lane)*16 + j ; B[k][n], n = nt*16 + (lane&15),
// k = kt*32 + (lane>>4)*16 + j.  W is (Cout, Cin=64) row-major (o-major).
__global__ void pack_frag_kernel(const float* __restrict__ W,
                                 unsigned short* __restrict__ frag, int Cout) {
  int t = blockIdx.x * blockDim.x + threadIdx.x;
  int total = (Cout >> 4) * 2 * 32 * 16;
  if (t >= total) return;
  int j = t & 15, lane = (t >> 4) & 31, kt = (t >> 9) & 1, nt = t >> 10;
  int n = nt * 16 + (lane & 15);
  int k = kt * 32 + (lane >> 4) * 16 + j;
  frag[t] = f2bf(W[n * 64 + k]);
}

// ---------------- fc0: (x, gridx, gridy) -> 64 channels ----------------
__global__ void fc0_kernel(const float* __restrict__ x,
                           const float* __restrict__ w,
                           const float* __restrict__ bvec,
                           float* __restrict__ out) {
  int t = blockIdx.x * blockDim.x + threadIdx.x;
  if (t >= NB * CC * HWP) return;
  int p = t % HWP; int c = (t / HWP) % CC; int b = t / (HWP * CC);
  int py = p >> 7, px = p & 127;
  float gx = py * (1.0f / 127.0f);
  float gy = px * (1.0f / 127.0f);
  float xv = x[(long)b * HWP + p];
  out[t] = bvec[c] + w[c * 3 + 0] * xv + w[c * 3 + 1] * gx + w[c * 3 + 2] * gy;
}

// ---------------- generic 1x1-conv WMMA GEMM (Cin=64, Cout=64) ----------------
// 4 waves/block, one 16-pixel tile per wave; out = act(bias + in @ W [+ addsrc])
template <bool HASADD, bool MISH>
__global__ __launch_bounds__(128) void gemm16_t(
    const float* __restrict__ in, const unsigned short* __restrict__ wfrag,
    const float* __restrict__ bias, const float* __restrict__ addsrc,
    float* __restrict__ out) {
  __shared__ __attribute__((aligned(16))) unsigned short lds[4][16 * 64];
  int lane = threadIdx.x & 31, wv = threadIdx.x >> 5;
  int tile = blockIdx.x * 4 + wv;
  int b = tile >> 10;                       // 1024 tiles per batch image
  int p0 = (tile & 1023) << 4;
  unsigned short* ldsA = lds[wv];
  stageA<HWP>(in + (size_t)b * 64 * HWP + p0, ldsA, lane);
  __syncthreads();
  int m = lane & 15, hi = lane >> 4;
  const v8bf* rowA = (const v8bf*)(ldsA + m * 64);
  ABfrag a0, a1;
  a0.h[0] = rowA[0 + 2 * hi]; a0.h[1] = rowA[1 + 2 * hi];
  a1.h[0] = rowA[4 + 2 * hi]; a1.h[1] = rowA[5 + 2 * hi];
#pragma unroll
  for (int nt = 0; nt < 4; ++nt) {
    const v8bf* wp0 = (const v8bf*)(wfrag + (((nt * 2 + 0) * 32 + lane) << 4));
    const v8bf* wp1 = (const v8bf*)(wfrag + (((nt * 2 + 1) * 32 + lane) << 4));
    ABfrag b0, b1;
    b0.h[0] = wp0[0]; b0.h[1] = wp0[1];
    b1.h[0] = wp1[0]; b1.h[1] = wp1[1];
    v8f acc = {};
    acc = __builtin_amdgcn_wmma_f32_16x16x32_bf16(false, a0.v, false, b0.v,
                                                  (short)0, acc, false, false);
    acc = __builtin_amdgcn_wmma_f32_16x16x32_bf16(false, a1.v, false, b1.v,
                                                  (short)0, acc, false, false);
    int n = nt * 16 + m;
    float bv = bias[n];
    size_t ob = (size_t)(b * 64 + n) * HWP + p0 + hi * 8;
    float v[8];
    if (HASADD) {
      const float4 A0 = *(const float4*)(addsrc + ob);
      const float4 A1 = *(const float4*)(addsrc + ob + 4);
      v[0] = acc[0] + bv + A0.x; v[1] = acc[1] + bv + A0.y;
      v[2] = acc[2] + bv + A0.z; v[3] = acc[3] + bv + A0.w;
      v[4] = acc[4] + bv + A1.x; v[5] = acc[5] + bv + A1.y;
      v[6] = acc[6] + bv + A1.z; v[7] = acc[7] + bv + A1.w;
    } else {
#pragma unroll
      for (int r = 0; r < 8; ++r) v[r] = acc[r] + bv;
    }
    if (MISH) {
#pragma unroll
      for (int r = 0; r < 8; ++r) v[r] = mishf(v[r]);
    }
    float4 o0 = {v[0], v[1], v[2], v[3]};
    float4 o1 = {v[4], v[5], v[6], v[7]};
    *(float4*)(out + ob) = o0;
    *(float4*)(out + ob + 4) = o1;
  }
}

// ---------------- coarsest-band mix: per (b,band) 64x64x64 GEMM ----------------
__global__ __launch_bounds__(128) void coarse_gemm_kernel(
    const float* __restrict__ cin, const unsigned short* __restrict__ mfrag,
    float* __restrict__ cout) {
  __shared__ __attribute__((aligned(16))) unsigned short lds[4][16 * 64];
  int lane = threadIdx.x & 31, wv = threadIdx.x >> 5;
  int tile = blockIdx.x * 4 + wv;           // 256 tiles total
  int mtile = tile & 3;
  int band  = (tile >> 2) & 3;
  int b     = tile >> 4;
  const float* in  = cin  + (size_t)band * (NB * 64 * 64);
  float*       out = cout + (size_t)band * (NB * 64 * 64);
  const unsigned short* wf = mfrag + (size_t)(b * 4 + band) * 4096;
  int p0 = mtile * 16;
  unsigned short* ldsA = lds[wv];
  stageA<64>(in + (size_t)b * 64 * 64 + p0, ldsA, lane);
  __syncthreads();
  int m = lane & 15, hi = lane >> 4;
  const v8bf* rowA = (const v8bf*)(ldsA + m * 64);
  ABfrag a0, a1;
  a0.h[0] = rowA[0 + 2 * hi]; a0.h[1] = rowA[1 + 2 * hi];
  a1.h[0] = rowA[4 + 2 * hi]; a1.h[1] = rowA[5 + 2 * hi];
#pragma unroll
  for (int nt = 0; nt < 4; ++nt) {
    const v8bf* wp0 = (const v8bf*)(wf + (((nt * 2 + 0) * 32 + lane) << 4));
    const v8bf* wp1 = (const v8bf*)(wf + (((nt * 2 + 1) * 32 + lane) << 4));
    ABfrag b0, b1;
    b0.h[0] = wp0[0]; b0.h[1] = wp0[1];
    b1.h[0] = wp1[0]; b1.h[1] = wp1[1];
    v8f acc = {};
    acc = __builtin_amdgcn_wmma_f32_16x16x32_bf16(false, a0.v, false, b0.v,
                                                  (short)0, acc, false, false);
    acc = __builtin_amdgcn_wmma_f32_16x16x32_bf16(false, a1.v, false, b1.v,
                                                  (short)0, acc, false, false);
    int n = nt * 16 + m;
    size_t ob = (size_t)(b * 64 + n) * 64 + p0 + hi * 8;
    float4 o0 = {acc[0], acc[1], acc[2], acc[3]};
    float4 o1 = {acc[4], acc[5], acc[6], acc[7]};
    *(float4*)(out + ob) = o0;
    *(float4*)(out + ob + 4) = o1;
  }
}

// ---------------- spatial mean over HxW ----------------
__global__ __launch_bounds__(256) void pool_kernel(const float* __restrict__ h,
                                                   float* __restrict__ pooled) {
  __shared__ float red[256];
  long bc = blockIdx.x;
  const float* p = h + bc * HWP;
  float s = 0.0f;
  for (int i = threadIdx.x; i < HWP; i += 256) s += p[i];
  red[threadIdx.x] = s;
  __syncthreads();
  for (int st = 128; st > 0; st >>= 1) {
    if ((int)threadIdx.x < st) red[threadIdx.x] += red[threadIdx.x + st];
    __syncthreads();
  }
  if (threadIdx.x == 0) pooled[bc] = red[0] * (1.0f / HWP);
}

// ---------------- gates: softmax over 8 experts, folded to 6 ----------------
__global__ void gates_kernel(const float* __restrict__ pooled,
                             const float* __restrict__ label,
                             const float* __restrict__ Wg,
                             const float* __restrict__ vg, int hd,
                             float* __restrict__ wts) {
  int t = blockIdx.x * blockDim.x + threadIdx.x;
  if (t >= NB * 64) return;
  float pv = pooled[t];
  float lg[8], mx = -1e30f;
#pragma unroll
  for (int e = 0; e < 8; ++e) {
    float g0 = 0.0f;
#pragma unroll
    for (int l = 0; l < 8; ++l) g0 += label[l] * Wg[(hd * 8 + l) * 8 + e];
    lg[e] = g0 + pv * vg[hd * 8 + e];
    mx = fmaxf(mx, lg[e]);
  }
  float sum = 0.0f;
#pragma unroll
  for (int e = 0; e < 8; ++e) { lg[e] = __expf(lg[e] - mx); sum += lg[e]; }
  float inv = 1.0f / sum;
  float* o = wts + (long)t * 6;
  o[0] = lg[0] * inv; o[1] = lg[1] * inv; o[2] = lg[2] * inv; o[3] = lg[3] * inv;
  o[4] = (lg[4] + lg[6]) * inv; o[5] = (lg[5] + lg[7]) * inv;
}

// ---------------- effective mix matrices (gate folded into weights) ----------------
// Meff[b,band][k][n] = sum_e wts[b,n,e] * exp_w[hd,e,band,k,n]  -> bf16 fragments
__global__ void meff_kernel(const float* __restrict__ wts,
                            const float* __restrict__ exp_w, int hd,
                            unsigned short* __restrict__ mfrag) {
  int t = blockIdx.x * blockDim.x + threadIdx.x;
  if (t >= NB * 4 * 4096) return;
  int j = t & 15, lane = (t >> 4) & 31, kt = (t >> 9) & 1;
  int nt = (t >> 10) & 3, band = (t >> 12) & 3, b = t >> 14;
  int n = nt * 16 + (lane & 15);
  int k = kt * 32 + (lane >> 4) * 16 + j;
  const float* wv = wts + (long)(b * 64 + n) * 6;
  float acc = 0.0f;
#pragma unroll
  for (int e = 0; e < 6; ++e)
    acc += wv[e] * exp_w[((long)((hd * 8 + e) * 4 + band) * 64 + k) * 64 + n];
  mfrag[t] = f2bf(acc);
}

// ---------------- one Haar DWT level ----------------
__global__ void dwt_kernel(const float* __restrict__ in, float* __restrict__ ll,
                           float* __restrict__ lh, float* __restrict__ hl,
                           float* __restrict__ hh, int half) {
  int t = blockIdx.x * blockDim.x + threadIdx.x;
  if (t >= NB * 64 * half * half) return;
  int x = t % half, y = (t / half) % half, ch = t / (half * half);
  int full = half * 2;
  const float* base = in + (long)ch * full * full;
  float a = base[(2 * y) * full + 2 * x];
  float b = base[(2 * y + 1) * full + 2 * x];
  float c = base[(2 * y) * full + 2 * x + 1];
  float d = base[(2 * y + 1) * full + 2 * x + 1];
  ll[t] = (a + b + c + d) * 0.5f;
  lh[t] = (a + b - c - d) * 0.5f;
  hl[t] = (a - b + c - d) * 0.5f;
  hh[t] = (a - b - c + d) * 0.5f;
}

// ---------------- one inverse Haar DWT level ----------------
__global__ void idwt_kernel(const float* __restrict__ ll,
                            const float* __restrict__ lh,
                            const float* __restrict__ hl,
                            const float* __restrict__ hh,
                            float* __restrict__ out, int half) {
  int t = blockIdx.x * blockDim.x + threadIdx.x;
  if (t >= NB * 64 * half * half) return;
  int x = t % half, y = (t / half) % half, ch = t / (half * half);
  int full = half * 2;
  float A = ll[t], B = lh[t], C = hl[t], D = hh[t];
  float* base = out + (long)ch * full * full;
  base[(2 * y) * full + 2 * x]         = (A + B + C + D) * 0.5f;
  base[(2 * y) * full + 2 * x + 1]     = (A - B + C - D) * 0.5f;
  base[(2 * y + 1) * full + 2 * x]     = (A + B - C - D) * 0.5f;
  base[(2 * y + 1) * full + 2 * x + 1] = (A - B - C + D) * 0.5f;
}

// ---------------- fused fc2 (64->128) + mish + fc3 (128->1) ----------------
__global__ __launch_bounds__(128) void fc2fc3_kernel(
    const float* __restrict__ in, const unsigned short* __restrict__ wfrag,
    const float* __restrict__ fc2b, const float* __restrict__ fc3w,
    const float* __restrict__ fc3b, float* __restrict__ out) {
  __shared__ __attribute__((aligned(16))) unsigned short lds[4][16 * 64];
  __shared__ float ldsO[4][16];
  int lane = threadIdx.x & 31, wv = threadIdx.x >> 5;
  int tile = blockIdx.x * 4 + wv;
  int b = tile >> 10;
  int p0 = (tile & 1023) << 4;
  unsigned short* ldsA = lds[wv];
  stageA<HWP>(in + (size_t)b * 64 * HWP + p0, ldsA, lane);
  if (lane < 16) ldsO[wv][lane] = 0.0f;
  __syncthreads();
  int m = lane & 15, hi = lane >> 4;
  const v8bf* rowA = (const v8bf*)(ldsA + m * 64);
  ABfrag a0, a1;
  a0.h[0] = rowA[0 + 2 * hi]; a0.h[1] = rowA[1 + 2 * hi];
  a1.h[0] = rowA[4 + 2 * hi]; a1.h[1] = rowA[5 + 2 * hi];
  float psum[8];
#pragma unroll
  for (int r = 0; r < 8; ++r) psum[r] = 0.0f;
#pragma unroll
  for (int nt = 0; nt < 8; ++nt) {
    const v8bf* wp0 = (const v8bf*)(wfrag + (((nt * 2 + 0) * 32 + lane) << 4));
    const v8bf* wp1 = (const v8bf*)(wfrag + (((nt * 2 + 1) * 32 + lane) << 4));
    ABfrag b0, b1;
    b0.h[0] = wp0[0]; b0.h[1] = wp0[1];
    b1.h[0] = wp1[0]; b1.h[1] = wp1[1];
    v8f acc = {};
    acc = __builtin_amdgcn_wmma_f32_16x16x32_bf16(false, a0.v, false, b0.v,
                                                  (short)0, acc, false, false);
    acc = __builtin_amdgcn_wmma_f32_16x16x32_bf16(false, a1.v, false, b1.v,
                                                  (short)0, acc, false, false);
    int n = nt * 16 + m;
    float bb = fc2b[n], w3 = fc3w[n];
#pragma unroll
    for (int r = 0; r < 8; ++r) psum[r] += w3 * mishf(acc[r] + bb);
  }
#pragma unroll
  for (int r = 0; r < 8; ++r) atomicAdd(&ldsO[wv][hi * 8 + r], psum[r]);
  __syncthreads();
  if (lane < 16) out[(size_t)b * HWP + p0 + lane] = ldsO[wv][lane] + fc3b[0];
}

// ---------------- host orchestration ----------------
extern "C" void kernel_launch(void* const* d_in, const int* in_sizes, int n_in,
                              void* d_out, int out_size, void* d_ws, size_t ws_size,
                              hipStream_t stream) {
  (void)in_sizes; (void)n_in; (void)out_size; (void)ws_size;
  const float* x     = (const float*)d_in[0];
  const float* label = (const float*)d_in[1];
  const float* fc0_w = (const float*)d_in[2];
  const float* fc0_b = (const float*)d_in[3];
  const float* fc1_w = (const float*)d_in[4];
  const float* fc1_b = (const float*)d_in[5];
  const float* gWg   = (const float*)d_in[6];
  const float* gvg   = (const float*)d_in[7];
  const float* exp_w = (const float*)d_in[8];
  const float* w0_w  = (const float*)d_in[9];
  const float* w0_b  = (const float*)d_in[10];
  const float* fc2_w = (const float*)d_in[11];
  const float* fc2_b = (const float*)d_in[12];
  const float* fc3_w = (const float*)d_in[13];
  const float* fc3_b = (const float*)d_in[14];
  float* outp = (float*)d_out;

  char* wsb = (char*)d_ws;
  size_t off = 0;
  auto alloc = [&](size_t bytes) -> char* {
    char* p = wsb + off;
    off = (off + bytes + 255) & ~(size_t)255;
    return p;
  };
  const size_t FULL = (size_t)NB * CC * HWP;         // 16.78M floats
  float* hA   = (float*)alloc(FULL * 4);
  float* hB   = (float*)alloc(FULL * 4);
  float* Y    = (float*)alloc(FULL * 4);
  float* S1   = (float*)alloc((FULL / 4) * 4);       // 64x64 LL / recon
  float* S2   = (float*)alloc((FULL / 16) * 4);      // 32x32
  float* S3   = (float*)alloc((FULL / 64) * 4);      // 16x16
  float* D1   = (float*)alloc(3 * (FULL / 4) * 4);   // lh,hl,hh @64
  float* D2   = (float*)alloc(3 * (FULL / 16) * 4);
  float* D3   = (float*)alloc(3 * (FULL / 64) * 4);
  float* C4i  = (float*)alloc(4 * (FULL / 256) * 4); // [A4, lh4, hl4, hh4] @8
  float* C4o  = (float*)alloc(4 * (FULL / 256) * 4); // mixed bands
  float* pooled = (float*)alloc(NB * 64 * 4);
  float* wts    = (float*)alloc(NB * 64 * 6 * 4);
  unsigned short* fc1f = (unsigned short*)alloc(4096 * 2);
  unsigned short* w0f  = (unsigned short*)alloc(4 * 4096 * 2);
  unsigned short* fc2f = (unsigned short*)alloc(8192 * 2);
  unsigned short* mef  = (unsigned short*)alloc((size_t)NB * 4 * 4096 * 2);

  const size_t d1 = FULL / 4, d2 = FULL / 16, d3 = FULL / 64, d4 = FULL / 256;

  // pack static weights to bf16 fragments
  pack_frag_kernel<<<16, 256, 0, stream>>>(fc1_w, fc1f, 64);
  for (int hd = 0; hd < HID; ++hd)
    pack_frag_kernel<<<16, 256, 0, stream>>>(w0_w + (size_t)hd * 4096,
                                             w0f + (size_t)hd * 4096, 64);
  pack_frag_kernel<<<32, 256, 0, stream>>>(fc2_w, fc2f, 128);

  // lifting: fc0 then fc1
  fc0_kernel<<<(int)(FULL / 256), 256, 0, stream>>>(x, fc0_w, fc0_b, hB);
  gemm16_t<false, false><<<NB * 256, 128, 0, stream>>>(hB, fc1f, fc1_b, nullptr, hA);

  float* cur = hA;
  float* nxt = hB;
  for (int hd = 0; hd < HID; ++hd) {
    // gates
    pool_kernel<<<NB * 64, 256, 0, stream>>>(cur, pooled);
    gates_kernel<<<4, 256, 0, stream>>>(pooled, label, gWg, gvg, hd, wts);
    meff_kernel<<<(int)(NB * 4 * 4096 / 256), 256, 0, stream>>>(wts, exp_w, hd, mef);
    // forward DWT pyramid (details of levels 1-3 pass through since sum(gates)==1)
    dwt_kernel<<<(int)(d1 / 256), 256, 0, stream>>>(cur, S1, D1, D1 + d1, D1 + 2 * d1, 64);
    dwt_kernel<<<(int)(d2 / 256), 256, 0, stream>>>(S1, S2, D2, D2 + d2, D2 + 2 * d2, 32);
    dwt_kernel<<<(int)(d3 / 256), 256, 0, stream>>>(S2, S3, D3, D3 + d3, D3 + 2 * d3, 16);
    dwt_kernel<<<(int)(d4 / 256), 256, 0, stream>>>(S3, C4i, C4i + d4, C4i + 2 * d4,
                                                    C4i + 3 * d4, 8);
    // gate-folded channel mix of the coarsest bands (WMMA, per-batch matrices)
    coarse_gemm_kernel<<<64, 128, 0, stream>>>(C4i, mef, C4o);
    // inverse DWT
    idwt_kernel<<<(int)(d4 / 256), 256, 0, stream>>>(C4o, C4o + d4, C4o + 2 * d4,
                                                     C4o + 3 * d4, S3, 8);
    idwt_kernel<<<(int)(d3 / 256), 256, 0, stream>>>(S3, D3, D3 + d3, D3 + 2 * d3, S2, 16);
    idwt_kernel<<<(int)(d2 / 256), 256, 0, stream>>>(S2, D2, D2 + d2, D2 + 2 * d2, S1, 32);
    idwt_kernel<<<(int)(d1 / 256), 256, 0, stream>>>(S1, D1, D1 + d1, D1 + 2 * d1, Y, 64);
    // skip conv + residual add + mish (except last layer)
    if (hd != HID - 1)
      gemm16_t<true, true><<<NB * 256, 128, 0, stream>>>(
          cur, w0f + (size_t)hd * 4096, w0_b + (size_t)hd * 64, Y, nxt);
    else
      gemm16_t<true, false><<<NB * 256, 128, 0, stream>>>(
          cur, w0f + (size_t)hd * 4096, w0_b + (size_t)hd * 64, Y, nxt);
    float* t = cur; cur = nxt; nxt = t;
  }

  // projection head
  fc2fc3_kernel<<<NB * 256, 128, 0, stream>>>(cur, fc2f, fc2_b, fc3_w, fc3_b, outp);
}
